// TransducerJoint_20409684591206
// MI455X (gfx1250) — compile-verified, matrix operand order checked by reading, not run on hardware
//
#include <hip/hip_runtime.h>
#include <hip/hip_bf16.h>

// Problem sizes (fixed by the reference)
#define Bdim 8
#define Tdim 200
#define Udim 100
#define Edim 512
#define Pdim 640
#define Hdim 512
#define Vdim 1024
#define Mtot (Bdim * Tdim * Udim)   // 160000 flattened joint rows
#define MTILE 64                    // rows per block (160000 = 64 * 2500)

typedef __bf16 bf16_t;
typedef __attribute__((ext_vector_type(16))) __bf16 v16bf;
typedef __attribute__((ext_vector_type(8)))  __bf16 v8bf;
typedef __attribute__((ext_vector_type(8)))  float  v8f;
typedef __attribute__((ext_vector_type(4)))  float  v4f;

// ---------------------------------------------------------------------------
// Small f32 projection: Y[row,h] = sum_e X[row,e]*W[h,e] + bias[h]
// ---------------------------------------------------------------------------
template <int IN>
__global__ __launch_bounds__(256) void proj_kernel(
    const float* __restrict__ X,    // (rows, IN)
    const float* __restrict__ W,    // (H, IN) row-major
    const float* __restrict__ bias, // (H)
    float* __restrict__ Y)          // (rows, H)
{
  __shared__ float xs[IN];
  const int row = blockIdx.x;
  const float* xr = X + (size_t)row * IN;
  for (int i = threadIdx.x; i < IN; i += 256) xs[i] = xr[i];
  __syncthreads();

  for (int h = threadIdx.x; h < Hdim; h += 256) {
    const float* wr = W + (size_t)h * IN;
    float s = 0.f;
#pragma unroll 4
    for (int e = 0; e < IN; e += 4) {
      v4f w4 = *(const v4f*)(wr + e);
      v4f x4 = *(const v4f*)(&xs[e]);
      s += w4[0] * x4[0] + w4[1] * x4[1] + w4[2] * x4[2] + w4[3] * x4[3];
    }
    Y[(size_t)row * Hdim + h] = s + bias[h];
  }
}

// ---------------------------------------------------------------------------
// f32 -> bf16 conversion (W_out), RNE
// ---------------------------------------------------------------------------
__global__ __launch_bounds__(256) void cvt_bf16_kernel(
    const float* __restrict__ src, bf16_t* __restrict__ dst, int n)
{
  int i = (blockIdx.x * 256 + threadIdx.x) * 4;
  if (i + 3 < n) {
    v4f x = *(const v4f*)(src + i);
    dst[i + 0] = (bf16_t)x[0];
    dst[i + 1] = (bf16_t)x[1];
    dst[i + 2] = (bf16_t)x[2];
    dst[i + 3] = (bf16_t)x[3];
  }
}

// ---------------------------------------------------------------------------
// Main fused kernel: block = 64 flattened (b,t,u) rows x full V=1024.
//   A[m,k] = bf16(relu(encP[row] + predP[row]))  built in 64KB swizzled LDS
//   D = A(64x512) x W_out^T(512x1024), +b_out, *0.1 -> out
// 16 waves (512 thr); each wave: 64-wide N strip x all 64 M rows
//   => per K-step: 4 A frags (LDS) + 4 B frags (global) -> 16 WMMAs.
// LDS XOR swizzle: addr(row,k) = row*512 + (((k>>3) ^ (row&15)) << 3) + (k&7)
// => conflict-free ds_load_b128 without padding (exactly 64KB).
// ---------------------------------------------------------------------------
__device__ __forceinline__ int lds_chunk_addr(int row, int chunk) {
  return row * Hdim + ((chunk ^ (row & 15)) << 3);
}

__global__ __launch_bounds__(512) void joint_gemm_kernel(
    const float* __restrict__ encP,   // (B*T, H) f32
    const float* __restrict__ predP,  // (B*U, H) f32
    const bf16_t* __restrict__ Wb,    // (V, H) bf16, row-major
    const float* __restrict__ b_out,  // (V)
    float* __restrict__ out)          // (B*T*U, V) f32
{
  __shared__ bf16_t Atile[MTILE * Hdim];  // 64 KB, swizzled

  const int fr0 = blockIdx.x * MTILE;  // flattened ((b*T+t)*U+u) base row
  const int tid = threadIdx.x;

  // ---- build 64 x 512 bf16 A tile (8 threads per row, 64 k each) ----
  {
    const int row = tid >> 3;        // 0..63
    const int j   = tid & 7;         // 0..7
    const int fr  = fr0 + row;
    // fr = (b*T+t)*U + u  =>  enc row = fr/U ; pred row = (fr/(T*U))*U + fr%U
    const int encRow  = fr / Udim;
    const int predRow = (fr / (Tdim * Udim)) * Udim + (fr % Udim);
    const float* er = encP  + (size_t)encRow  * Hdim;
    const float* pr = predP + (size_t)predRow * Hdim;
    const int k0 = j * 64;
#pragma unroll
    for (int kc = 0; kc < 64; kc += 8) {
      const int k = k0 + kc;
      v4f e0 = *(const v4f*)(er + k);
      v4f e1 = *(const v4f*)(er + k + 4);
      v4f p0 = *(const v4f*)(pr + k);
      v4f p1 = *(const v4f*)(pr + k + 4);
      v8bf w;
#pragma unroll
      for (int q = 0; q < 4; ++q) {
        float s0 = e0[q] + p0[q]; s0 = s0 > 0.f ? s0 : 0.f;
        float s1 = e1[q] + p1[q]; s1 = s1 > 0.f ? s1 : 0.f;
        w[q]     = (bf16_t)s0;
        w[4 + q] = (bf16_t)s1;
      }
      *(v8bf*)&Atile[lds_chunk_addr(row, k >> 3)] = w;
    }
  }
  __syncthreads();

  const int wave = tid >> 5;          // 0..15 (wave32)
  const int lane = tid & 31;
  const int rc   = lane & 15;         // A: M row low bits / B: N col in tile
  const int klo  = (lane >> 4) << 3;  // 0 or 8 (CDNA5 16-bit frag K offset)
  const int nBase = wave * 64;        // this wave's V strip (64 wide)

  v8f acc[4][4] = {};

  // Per-ni B row pointers (v = nBase + ni*16 + rc), row-major (V,H)
  const bf16_t* Brow0 = Wb + (size_t)(nBase + rc) * Hdim;

#pragma unroll 1
  for (int kb = 0; kb < Hdim; kb += 32) {
    // ---- 4 A fragments from swizzled LDS ----
    v16bf a[4];
#pragma unroll
    for (int mi = 0; mi < 4; ++mi) {
      const int row = mi * 16 + rc;
      const int c0 = (kb + klo) >> 3;        // chunk idx pre-swizzle
      v8bf a0 = *(const v8bf*)&Atile[lds_chunk_addr(row, c0)];
      v8bf a1 = *(const v8bf*)&Atile[lds_chunk_addr(row, c0 + 2)];
#pragma unroll
      for (int q = 0; q < 8; ++q) { a[mi][q] = a0[q]; a[mi][8 + q] = a1[q]; }
    }

    // ---- 4 B fragments, each reused by 4 WMMAs ----
#pragma unroll
    for (int ni = 0; ni < 4; ++ni) {
      const bf16_t* Brow = Brow0 + (size_t)(ni * 16) * Hdim + kb;
      v8bf b0 = *(const v8bf*)(Brow + klo);
      v8bf b1 = *(const v8bf*)(Brow + 16 + klo);
      v16bf bb;
#pragma unroll
      for (int q = 0; q < 8; ++q) { bb[q] = b0[q]; bb[8 + q] = b1[q]; }

#pragma unroll
      for (int mi = 0; mi < 4; ++mi) {
        acc[mi][ni] = __builtin_amdgcn_wmma_f32_16x16x32_bf16(
            false, a[mi], false, bb, (short)0, acc[mi][ni], false, false);
      }
    }
  }

  // ---- epilogue: lane holds n=rc; acc VGPR r holds m = r + 8*(lane>>4) ----
  const int mhalf = (lane >> 4) * 8;
#pragma unroll
  for (int ni = 0; ni < 4; ++ni) {
    const int v = nBase + ni * 16 + rc;
    const float bo = b_out[v];
#pragma unroll
    for (int mi = 0; mi < 4; ++mi) {
#pragma unroll
      for (int r = 0; r < 8; ++r) {
        const int m = mi * 16 + mhalf + r;
        float* p = out + (size_t)(fr0 + m) * Vdim + v;  // out contiguous in fr
        __builtin_nontemporal_store(0.1f * (acc[mi][ni][r] + bo), p);
      }
    }
  }
}

// ---------------------------------------------------------------------------
extern "C" void kernel_launch(void* const* d_in, const int* in_sizes, int n_in,
                              void* d_out, int out_size, void* d_ws, size_t ws_size,
                              hipStream_t stream) {
  (void)in_sizes; (void)n_in; (void)out_size; (void)ws_size;

  const float* enc    = (const float*)d_in[0];  // (B,T,E)
  const float* pred   = (const float*)d_in[1];  // (B,U,P)
  const float* W_enc  = (const float*)d_in[2];  // (H,E)
  const float* b_enc  = (const float*)d_in[3];  // (H)
  const float* W_pred = (const float*)d_in[4];  // (H,P)
  const float* b_pred = (const float*)d_in[5];  // (H)
  const float* W_out  = (const float*)d_in[6];  // (V,H)
  const float* b_out  = (const float*)d_in[7];  // (V)
  float* out = (float*)d_out;

  // workspace layout
  float*  encP  = (float*)d_ws;                          // B*T*H f32  (3.28 MB)
  float*  predP = encP + (size_t)Bdim * Tdim * Hdim;     // B*U*H f32  (1.64 MB)
  bf16_t* Wb    = (bf16_t*)(predP + (size_t)Bdim * Udim * Hdim); // V*H bf16 (1 MB)

  proj_kernel<Edim><<<Bdim * Tdim, 256, 0, stream>>>(enc, W_enc, b_enc, encP);
  proj_kernel<Pdim><<<Bdim * Udim, 256, 0, stream>>>(pred, W_pred, b_pred, predP);
  cvt_bf16_kernel<<<(Vdim * Hdim / 4 + 255) / 256, 256, 0, stream>>>(W_out, Wb, Vdim * Hdim);

  joint_gemm_kernel<<<Mtot / MTILE, 512, 0, stream>>>(encP, predP, Wb, b_out, out);
}
